// FMoE_76381698392953
// MI455X (gfx1250) — compile-verified, matrix operand order checked by reading, not run on hardware
//
#include <hip/hip_runtime.h>
#include <hip/hip_bf16.h>

// ---- problem constants (match reference) ----
#define N_TOK 4096
#define DMODEL 1024
#define NEXP 16
#define TOPK 2
#define HID 2048
#define CAPC 1024

typedef unsigned short u16;
typedef unsigned int u32;
typedef __attribute__((ext_vector_type(16))) __bf16 v16bf;
typedef __attribute__((ext_vector_type(2)))  __bf16 v2bf;
typedef __attribute__((ext_vector_type(8)))  float  v8f;
typedef __attribute__((ext_vector_type(2)))  float  v2f;
typedef __attribute__((ext_vector_type(4)))  u32    u32x4;
typedef __attribute__((ext_vector_type(4)))  u16    u16x4;

// float -> bf16 via native convert (RTNE); backend picks v_cvt_*_bf16_f32
__device__ __forceinline__ u16 f2bf(float f) {
    __bf16 h = (__bf16)f;
    return __builtin_bit_cast(u16, h);
}
// packed pair: gives backend a shot at v_cvt_pk_bf16_f32
__device__ __forceinline__ u32 f2bf2(float lo, float hi) {
    v2f f; f.x = lo; f.y = hi;
    v2bf h = __builtin_convertvector(f, v2bf);
    return __builtin_bit_cast(u32, h);
}

// tanh-approximate GELU via sigmoid identity: 0.5*(1+tanh(z)) == sigmoid(2z).
__device__ __forceinline__ float gelu_fast(float x) {
    const float k = 1.5957691216057308f;   // 2*sqrt(2/pi)
    float x3 = x * x * x;
    float z = k * (x + 0.044715f * x3);
    return x * __builtin_amdgcn_rcpf(1.0f + __expf(-z));
}

// Build a 16xbf16 WMMA fragment from LDS: 8 halves at +0, 8 at +16.
// ISA 16-bit A/B striping: lanes 0-15 -> K 0..7 & 16..23, lanes 16-31 -> K 8..15 & 24..31.
__device__ __forceinline__ v16bf ld_frag(const u16* row) {
    u32x4 lo = *(const u32x4*)(row);
    u32x4 hi = *(const u32x4*)(row + 16);
    union { u32x4 u[2]; v16bf v; } cvt;
    cvt.u[0] = lo; cvt.u[1] = hi;
    return cvt.v;
}

// ------------------------------------------------------------------
// 1) zero expert counters
__global__ void zero_counts_kernel(int* counts) {
    if (threadIdx.x < NEXP) counts[threadIdx.x] = 0;
}

// 2) gate logits: [N,E] = X @ Wg + bg
__global__ __launch_bounds__(256) void gate_logits_kernel(
    const float* __restrict__ X, const float* __restrict__ Wg,
    const float* __restrict__ bg, float* __restrict__ logits) {
    int idx = blockIdx.x * blockDim.x + threadIdx.x;   // N*E threads
    int n = idx >> 4, e = idx & 15;
    const float* xr = X + (size_t)n * DMODEL;
    float s = 0.f;
    #pragma unroll 4
    for (int d = 0; d < DMODEL; ++d) s += xr[d] * Wg[d * NEXP + e];
    logits[idx] = s + bg[e];
}

// 3) top-2 + softmax + slot assignment (slot order doesn't affect the output:
//    each row's WMMA result depends only on that row and the expert weights)
__global__ void gate_top2_kernel(const float* __restrict__ logits, int* counts,
                                 int* __restrict__ assign_e, int* __restrict__ assign_pos,
                                 float* __restrict__ gate_w) {
    int n = blockIdx.x * blockDim.x + threadIdx.x;
    if (n >= N_TOK) return;
    const float* l = logits + (size_t)n * NEXP;
    int i0 = 0; float v0 = l[0];
    #pragma unroll
    for (int e = 1; e < NEXP; ++e) { float v = l[e]; if (v > v0) { v0 = v; i0 = e; } }
    int i1 = -1; float v1 = -3.4e38f;
    #pragma unroll
    for (int e = 0; e < NEXP; ++e) { if (e == i0) continue; float v = l[e]; if (v > v1) { v1 = v; i1 = e; } }
    float ex = __expf(v1 - v0);
    float inv = 1.0f / (1.0f + ex);
    float g0 = inv, g1 = ex * inv;
    int p0 = atomicAdd(&counts[i0], 1);
    int p1 = atomicAdd(&counts[i1], 1);
    assign_e[2 * n] = i0;  assign_pos[2 * n] = p0;  gate_w[2 * n] = g0;
    assign_e[2 * n + 1] = i1; assign_pos[2 * n + 1] = p1; gate_w[2 * n + 1] = g1;
}

// 4) dispatch: copy token row -> x_e[e][pos] as bf16 (float4 in, 4xbf16 out)
__global__ __launch_bounds__(128) void dispatch_kernel(
    const float* __restrict__ X, const int* __restrict__ assign_e,
    const int* __restrict__ assign_pos, u16* __restrict__ x_e) {
    int a = blockIdx.x;                    // assignment id, N*K blocks
    int n = a >> 1;
    int e = assign_e[a], p = assign_pos[a];
    if (p >= CAPC) return;
    const float4* src = (const float4*)(X + (size_t)n * DMODEL);
    u32* dst = (u32*)(x_e + ((size_t)e * CAPC + p) * DMODEL);
    #pragma unroll
    for (int it = 0; it < 2; ++it) {
        int i = threadIdx.x + it * 128;    // 256 float4 chunks
        float4 f = src[i];
        dst[i * 2]     = f2bf2(f.x, f.y);
        dst[i * 2 + 1] = f2bf2(f.z, f.w);
    }
}

// ------------------------------------------------------------------
// Expert GEMM: C[E,CAP,NT] = act(A[E,CAP,KT](bf16) @ B[E,KT,NT](f32->bf16) + bias)
// 256 thr (8 waves), 128x128 C tile, K-step 32, double-buffered LDS,
// register-staged pipeline: one barrier/iter; uniform-base + precomputed
// per-thread offsets so stage loads avoid per-iteration address math.
template<int KT, int NT, bool GELU_BF16_OUT>
__global__ __launch_bounds__(256) void expert_gemm_kernel(
    const u16* __restrict__ Aall, const float* __restrict__ Ball,
    const float* __restrict__ biasAll, void* __restrict__ Call,
    const int* __restrict__ counts, int ntiles) {
    const int e     = blockIdx.y;
    const int mtile = blockIdx.x / ntiles;
    const int ntile = blockIdx.x % ntiles;
    int cnt = counts[e]; if (cnt > CAPC) cnt = CAPC;
    const int mbase = mtile * 128;
    if (mbase >= cnt) return;                       // skip empty expert tiles
    const int nbase = ntile * 128;

    const u16*   Abase = Aall    + (size_t)e * CAPC * KT + (size_t)mbase * KT;
    const float* Bbase = Ball    + (size_t)e * KT * NT + nbase;
    const float* bias  = biasAll + (size_t)e * NT;

    // rows padded to 40 halves (80B): 16B-aligned row starts, bank spread
    __shared__ u16 sA[2][128 * 40];   // [mLocal][k]  row-major
    __shared__ u16 sB[2][128 * 40];   // [nLocal][k]  (transposed: contiguous K per column)

    const int tid   = threadIdx.x;
    const int lane  = tid & 31;
    const int wid   = tid >> 5;
    const int wm    = wid >> 2;                 // 0..1  -> 64-row band
    const int wn    = wid & 3;                  // 0..3  -> 32-col band
    const int lrow  = lane & 15;
    const int khalf = (lane >> 4) << 3;         // 0 or 8

    // ---- precomputed per-thread offsets (loop-invariant) ----
    int aGoff[2], aLoff[2];                     // A: 512 16B-chunks, 2/thread
    #pragma unroll
    for (int it = 0; it < 2; ++it) {
        int chunk = tid + it * 256;
        int r = chunk >> 2, cc = chunk & 3;
        aGoff[it] = r * KT + cc * 8;            // elements (u16)
        aLoff[it] = r * 40 + cc * 8;
    }
    int bGoff[2], bLoff[2];                     // B: 512 (k-pair x 4col) tasks, 2/thread
    #pragma unroll
    for (int it = 0; it < 2; ++it) {
        int task = tid + it * 256;
        int kp = task >> 5, c4 = task & 31;     // k rows 2kp,2kp+1; cols c4*4..+3
        bGoff[it] = (2 * kp) * NT + c4 * 4;     // elements (float)
        bLoff[it] = (c4 * 4) * 20 + kp;         // u32 units (row = 20 u32)
    }
    // fragment LDS offsets (halves)
    int afOff[4], bfOff[2];
    #pragma unroll
    for (int i = 0; i < 4; ++i) afOff[i] = (wm * 64 + i * 16 + lrow) * 40 + khalf;
    #pragma unroll
    for (int j = 0; j < 2; ++j) bfOff[j] = (wn * 32 + j * 16 + lrow) * 40 + khalf;

    // staging registers for the in-flight stage
    u32x4  ra[2];
    float4 rb[4];

    auto load_stage = [&](int kk) {
        const u16*   ab = Abase + kk;
        const float* bb = Bbase + (size_t)kk * NT;
        #pragma unroll
        for (int it = 0; it < 2; ++it) ra[it] = *(const u32x4*)(ab + aGoff[it]);
        #pragma unroll
        for (int it = 0; it < 2; ++it) {
            rb[2 * it]     = *(const float4*)(bb + bGoff[it]);
            rb[2 * it + 1] = *(const float4*)(bb + bGoff[it] + NT);
        }
    };
    auto store_stage = [&](int buf) {
        u16* sa = sA[buf];
        u32* sb = (u32*)sB[buf];
        #pragma unroll
        for (int it = 0; it < 2; ++it) *(u32x4*)(sa + aLoff[it]) = ra[it];
        #pragma unroll
        for (int it = 0; it < 2; ++it) {
            float4 f0 = rb[2 * it], f1 = rb[2 * it + 1];
            int base = bLoff[it];
            sb[base +  0] = f2bf2(f0.x, f1.x);   // (n+0, k..k+1)
            sb[base + 20] = f2bf2(f0.y, f1.y);   // (n+1, ...)
            sb[base + 40] = f2bf2(f0.z, f1.z);
            sb[base + 60] = f2bf2(f0.w, f1.w);
        }
    };

    v8f acc[4][2];
    #pragma unroll
    for (int i = 0; i < 4; ++i)
        #pragma unroll
        for (int j = 0; j < 2; ++j) acc[i][j] = v8f{};

    constexpr int S = KT / 32;
    load_stage(0);
    store_stage(0);

    for (int s = 0; s < S; ++s) {
        __syncthreads();                           // buf (s&1) ready; prior compute drained
        if (s + 1 < S) load_stage((s + 1) * 32);   // VMEM in flight during WMMA below
        const int buf = s & 1;

        v16bf afr[4], bfr[2];
        #pragma unroll
        for (int i = 0; i < 4; ++i) afr[i] = ld_frag(&sA[buf][afOff[i]]);
        #pragma unroll
        for (int j = 0; j < 2; ++j) bfr[j] = ld_frag(&sB[buf][bfOff[j]]);

        #pragma unroll
        for (int i = 0; i < 4; ++i)
            #pragma unroll
            for (int j = 0; j < 2; ++j)
                acc[i][j] = __builtin_amdgcn_wmma_f32_16x16x32_bf16(
                    false, afr[i], false, bfr[j], (short)0, acc[i][j], false, false);

        if (s + 1 < S) store_stage(buf ^ 1);       // fill other buffer for next iter
    }

    // ---- epilogue: C layout: VGPR r -> M = r (+8 for lanes 16-31), N = lane&15 ----
    const int rowhal = (lane >> 4) << 3;
    #pragma unroll
    for (int j = 0; j < 2; ++j) {
        int col = nbase + wn * 32 + j * 16 + lrow;
        float bv = bias[col];
        #pragma unroll
        for (int i = 0; i < 4; ++i) {
            int rbase = mbase + wm * 64 + i * 16 + rowhal;
            #pragma unroll
            for (int r = 0; r < 8; ++r) {
                float v = acc[i][j][r] + bv;
                size_t idx = (size_t)e * CAPC * NT + (size_t)(rbase + r) * NT + col;
                if (GELU_BF16_OUT) ((u16*)Call)[idx]   = f2bf(gelu_fast(v));
                else               ((float*)Call)[idx] = v;
            }
        }
    }
}

// 7) gather + weighted combine (float4 throughout)
__global__ __launch_bounds__(256) void combine_kernel(
    const float* __restrict__ y_e, const int* __restrict__ assign_e,
    const int* __restrict__ assign_pos, const float* __restrict__ gate_w,
    float* __restrict__ out) {
    int n = blockIdx.x;
    int e0 = assign_e[2 * n], p0 = assign_pos[2 * n];
    int e1 = assign_e[2 * n + 1], p1 = assign_pos[2 * n + 1];
    float g0 = gate_w[2 * n], g1 = gate_w[2 * n + 1];
    bool ok0 = p0 < CAPC, ok1 = p1 < CAPC;
    const float4* y0 = (const float4*)(y_e + ((size_t)e0 * CAPC + (ok0 ? p0 : 0)) * DMODEL);
    const float4* y1 = (const float4*)(y_e + ((size_t)e1 * CAPC + (ok1 ? p1 : 0)) * DMODEL);
    float4* o = (float4*)(out + (size_t)n * DMODEL);
    int i = threadIdx.x;                       // 256 float4 chunks of 1024 floats
    float4 r = make_float4(0.f, 0.f, 0.f, 0.f);
    if (ok0) { float4 a = y0[i]; r.x += g0 * a.x; r.y += g0 * a.y; r.z += g0 * a.z; r.w += g0 * a.w; }
    if (ok1) { float4 b = y1[i]; r.x += g1 * b.x; r.y += g1 * b.y; r.z += g1 * b.z; r.w += g1 * b.w; }
    o[i] = r;
}

// ------------------------------------------------------------------
extern "C" void kernel_launch(void* const* d_in, const int* in_sizes, int n_in,
                              void* d_out, int out_size, void* d_ws, size_t ws_size,
                              hipStream_t stream) {
    const float* X  = (const float*)d_in[0];
    const float* Wg = (const float*)d_in[1];
    const float* bg = (const float*)d_in[2];
    const float* W1 = (const float*)d_in[3];
    const float* b1 = (const float*)d_in[4];
    const float* W2 = (const float*)d_in[5];
    const float* b2 = (const float*)d_in[6];
    float* out = (float*)d_out;

    char* ws = (char*)d_ws;
    size_t off = 0;
    auto alloc = [&](size_t bytes) -> void* {
        void* p = ws + off;
        off = (off + bytes + 255) & ~(size_t)255;
        return p;
    };
    int*   counts     = (int*)  alloc(NEXP * sizeof(int));
    float* logits     = (float*)alloc((size_t)N_TOK * NEXP * sizeof(float));
    int*   assign_e   = (int*)  alloc((size_t)N_TOK * TOPK * sizeof(int));
    int*   assign_pos = (int*)  alloc((size_t)N_TOK * TOPK * sizeof(int));
    float* gate_w     = (float*)alloc((size_t)N_TOK * TOPK * sizeof(float));
    u16*   x_e        = (u16*)  alloc((size_t)NEXP * CAPC * DMODEL * sizeof(u16));
    u16*   h_buf      = (u16*)  alloc((size_t)NEXP * CAPC * HID * sizeof(u16));
    float* y_e        = (float*)alloc((size_t)NEXP * CAPC * DMODEL * sizeof(float));

    zero_counts_kernel<<<1, 32, 0, stream>>>(counts);
    gate_logits_kernel<<<(N_TOK * NEXP) / 256, 256, 0, stream>>>(X, Wg, bg, logits);
    gate_top2_kernel<<<N_TOK / 128, 128, 0, stream>>>(logits, counts, assign_e, assign_pos, gate_w);
    dispatch_kernel<<<N_TOK * TOPK, 128, 0, stream>>>(X, assign_e, assign_pos, x_e);

    // GEMM1: [CAP,1024] x [1024,2048] -> gelu -> bf16 h
    {
        dim3 grid((CAPC / 128) * (HID / 128), NEXP);
        expert_gemm_kernel<DMODEL, HID, true><<<grid, 256, 0, stream>>>(
            x_e, W1, b1, (void*)h_buf, counts, HID / 128);
    }
    // GEMM2: [CAP,2048] x [2048,1024] -> f32 y_e
    {
        dim3 grid((CAPC / 128) * (DMODEL / 128), NEXP);
        expert_gemm_kernel<HID, DMODEL, false><<<grid, 256, 0, stream>>>(
            h_buf, W2, b2, (void*)y_e, counts, DMODEL / 128);
    }
    combine_kernel<<<N_TOK, 256, 0, stream>>>(y_e, assign_e, assign_pos, gate_w, out);
}
